// SelfAttention_87849261072676
// MI455X (gfx1250) — matrix operand order repaired and checked by reading, not compiled, not run on hardware
//
#include <hip/hip_runtime.h>
#include <hip/hip_bf16.h>

// ---------------------------------------------------------------------------
// CDNA5 (gfx1250) bf16 WMMA self-attention pipeline.
// wave32; WMMA shape 16x16x32 bf16 -> f32 accumulators.
//
// Fragment layout assumptions (CDNA5 ISA 7.12.2, wave32):
//   A (16x32 bf16): lane L holds row M = L&15; element e -> K = (e<8?e:e+8) + 8*(L>>4)
//   B (32x16 bf16): lane L holds col N = L&15; element e -> K = (e<8?e:e+8) + 8*(L>>4)
//   C (16x16 f32) : lane L, vgpr v ->  N = L&15 ; M = v + 8*(L>>4)
// ---------------------------------------------------------------------------

typedef __attribute__((ext_vector_type(16))) __bf16 v16bf;
typedef __attribute__((ext_vector_type(8)))  __bf16 bf16x8;
typedef __attribute__((ext_vector_type(8)))  float  v8f;

#define WMMA_BF16(a, b, c) \
  __builtin_amdgcn_wmma_f32_16x16x32_bf16(false, (a), false, (b), (short)0, (c), false, false)

static __device__ __forceinline__ bf16x8 ld8(const __bf16* p) {
  return *(const bf16x8*)p;
}
static __device__ __forceinline__ v16bf join8(bf16x8 lo, bf16x8 hi) {
  v16bf r;
#pragma unroll
  for (int i = 0; i < 8; ++i) { r[i] = lo[i]; r[i + 8] = hi[i]; }
  return r;
}
static __device__ __forceinline__ v8f zero8() {
  v8f z;
#pragma unroll
  for (int i = 0; i < 8; ++i) z[i] = 0.0f;
  return z;
}

// ---------------------------------------------------------------------------
// Kernel 0: convert f32 weights to bf16 in B-fragment-major layout.
// packed[((kt*NT + nt)*32 + lane)*16 + e] = W[kt*32 + (e&7) + 16*(e>>3) + 8*(lane>>4)][nt*16 + (lane&15)]
// ---------------------------------------------------------------------------
__global__ void pack_w_kernel(const float* __restrict__ W, __bf16* __restrict__ dst,
                              int Kd, int Nd) {
  const int NT = Nd >> 4;
  const int total = (Kd >> 5) * NT * 512;
  for (int idx = blockIdx.x * blockDim.x + threadIdx.x; idx < total;
       idx += gridDim.x * blockDim.x) {
    const int e    = idx & 15;
    const int lane = (idx >> 4) & 31;
    const int t    = idx >> 9;
    const int nt   = t % NT;
    const int kt   = t / NT;
    const int k = (kt << 5) + (e & 7) + ((e >> 3) << 4) + ((lane >> 4) << 3);
    const int n = (nt << 4) + (lane & 15);
    dst[idx] = (__bf16)W[(size_t)k * Nd + n];
  }
}

// ---------------------------------------------------------------------------
// Kernel 1: projection GEMM  [32768,1024] @ [1024,64] -> bf16, folded layout.
// grid.x = 512 (64 rows each), block = 128 (4 waves, 16 rows/wave).
// transposed==0 : dst[bh][sp][d]   (q, k)
// transposed==1 : dst[bh][d][sp]   (v, for contiguous B-fragments in P@V)
// ---------------------------------------------------------------------------
__global__ void __launch_bounds__(128)
proj_kernel(const float* __restrict__ X, const __bf16* __restrict__ pW,
            __bf16* __restrict__ dst, int transposed) {
  const int lane = threadIdx.x & 31;
  const int w    = threadIdx.x >> 5;
  const int h0   = lane >> 4;
  const int ln   = lane & 15;
  const int r0   = blockIdx.x * 64 + w * 16;

  v8f acc[4];
#pragma unroll
  for (int i = 0; i < 4; ++i) acc[i] = zero8();

  const float* xr = X + (size_t)(r0 + ln) * 1024;

  for (int kt = 0; kt < 32; ++kt) {
    const float* xp = xr + kt * 32;
    __builtin_prefetch(xp + 32, 0, 0);  // global_prefetch for next K-tile
    float4 x0 = *(const float4*)(xp + 8 * h0);
    float4 x1 = *(const float4*)(xp + 8 * h0 + 4);
    float4 x2 = *(const float4*)(xp + 16 + 8 * h0);
    float4 x3 = *(const float4*)(xp + 16 + 8 * h0 + 4);

    // Preload all four B fragments, then issue WMMAs (staggered loadcnt waits).
    v16bf bfr[4];
#pragma unroll
    for (int nt = 0; nt < 4; ++nt) {
      const __bf16* pf = pW + (((size_t)(kt * 4 + nt)) * 32 + lane) * 16;
      bfr[nt] = join8(ld8(pf), ld8(pf + 8));
    }
    v16bf a;
    a[0]  = (__bf16)x0.x; a[1]  = (__bf16)x0.y; a[2]  = (__bf16)x0.z; a[3]  = (__bf16)x0.w;
    a[4]  = (__bf16)x1.x; a[5]  = (__bf16)x1.y; a[6]  = (__bf16)x1.z; a[7]  = (__bf16)x1.w;
    a[8]  = (__bf16)x2.x; a[9]  = (__bf16)x2.y; a[10] = (__bf16)x2.z; a[11] = (__bf16)x2.w;
    a[12] = (__bf16)x3.x; a[13] = (__bf16)x3.y; a[14] = (__bf16)x3.z; a[15] = (__bf16)x3.w;
#pragma unroll
    for (int nt = 0; nt < 4; ++nt) acc[nt] = WMMA_BF16(a, bfr[nt], acc[nt]);
  }

#pragma unroll
  for (int vv = 0; vv < 8; ++vv) {
    const int gr = r0 + vv + 8 * h0;      // global row in [B*S)
    const int b_ = gr >> 13;              // / 8192
    const int s  = gr & 8191;
    const int sp = s >> 4;
    const int hh = s & 15;
    const int bh = b_ * 16 + hh;
#pragma unroll
    for (int nt = 0; nt < 4; ++nt) {
      const int n = nt * 16 + ln;
      const __bf16 val = (__bf16)acc[nt][vv];
      if (!transposed) dst[((size_t)bh * 512 + sp) * 64 + n] = val;
      else             dst[((size_t)bh * 64 + n) * 512 + sp] = val;
    }
  }
}

// ---------------------------------------------------------------------------
// Kernel 2: attention for one (b,h), 16 query rows per wave (1 wave / block).
// scores -> LDS (16 x 516 f32, padded), masked, two-pass softmax with deferred
// denominator, then P@V via WMMA.  grid = (32 qtiles, 64 bh), block = 32.
// ---------------------------------------------------------------------------
#define SROW 516

__global__ void __launch_bounds__(32)
attn_kernel(const __bf16* __restrict__ qh, const __bf16* __restrict__ kh,
            const __bf16* __restrict__ vt, const int* __restrict__ mask,
            __bf16* __restrict__ ao) {
  extern __shared__ float smem[];
  const int lane = threadIdx.x & 31;
  const int h0   = lane >> 4;
  const int ln   = lane & 15;
  const int bh   = blockIdx.y;
  const int b    = bh >> 4;
  const int h    = bh & 15;
  const int i0   = blockIdx.x * 16;

  float* S  = smem;               // [16][SROW]
  float* rs = smem + 16 * SROW;   // [16] row sums

  // Q A-fragments (row i0+ln, D=64 -> two K-steps), resident in registers.
  const __bf16* qrow = qh + ((size_t)bh * 512 + (i0 + ln)) * 64;
  const v16bf aq0 = join8(ld8(qrow + 8 * h0),      ld8(qrow + 16 + 8 * h0));
  const v16bf aq1 = join8(ld8(qrow + 32 + 8 * h0), ld8(qrow + 48 + 8 * h0));

  const int* mbase = mask + (size_t)bh * 512 * 512;

  // ---- pass 1: scores = Q K^T + mask * (-1e9) -> LDS ----
  for (int j0 = 0; j0 < 512; j0 += 16) {
    const __bf16* kb = kh + ((size_t)bh * 512 + (j0 + ln)) * 64;
    v16bf bk0 = join8(ld8(kb + 8 * h0),      ld8(kb + 16 + 8 * h0));
    v16bf bk1 = join8(ld8(kb + 32 + 8 * h0), ld8(kb + 48 + 8 * h0));
    // Two independent accumulator chains (no WMMA->WMMA RAW), combine in VALU.
    v8f c0 = zero8(), c1 = zero8();
    c0 = WMMA_BF16(aq0, bk0, c0);
    c1 = WMMA_BF16(aq1, bk1, c1);
    v8f c = c0 + c1;
#pragma unroll
    for (int vv = 0; vv < 8; ++vv) {
      const int i = i0 + vv + 8 * h0;
      const int m = mbase[(size_t)i * 512 + j0 + ln];
      S[(vv + 8 * h0) * SROW + j0 + ln] = c[vv] + (float)m * -1.0e9f;
    }
  }

  // ---- pass 2: rowwise unnormalized softmax (each half-wave owns 256 cols) ----
  {
    float* Sr = S + ln * SROW + h0 * 256;
    float mx = -3.0e38f;
#pragma unroll 4
    for (int c = 0; c < 256; c += 4) {
      float4 x = *(const float4*)(Sr + c);
      mx = fmaxf(mx, fmaxf(fmaxf(x.x, x.y), fmaxf(x.z, x.w)));
    }
    mx = fmaxf(mx, __shfl_xor(mx, 16, 32));
    float sm = 0.0f;
#pragma unroll 4
    for (int c = 0; c < 256; c += 4) {
      float4 x = *(float4*)(Sr + c);
      x.x = __expf(x.x - mx); x.y = __expf(x.y - mx);
      x.z = __expf(x.z - mx); x.w = __expf(x.w - mx);
      *(float4*)(Sr + c) = x;
      sm += x.x + x.y + x.z + x.w;
    }
    sm += __shfl_xor(sm, 16, 32);
    if (h0 == 0) rs[ln] = sm;
  }

  // ---- pass 3: O = P @ V (denominator deferred to epilogue) ----
  v8f acc[4];
#pragma unroll
  for (int i = 0; i < 4; ++i) acc[i] = zero8();

  for (int j0 = 0; j0 < 512; j0 += 32) {
    // Preload all four V B-fragments before the WMMA burst.
    v16bf bv[4];
#pragma unroll
    for (int nt = 0; nt < 4; ++nt) {
      const __bf16* vb = vt + ((size_t)bh * 64 + nt * 16 + ln) * 512 + j0 + 8 * h0;
      bv[nt] = join8(ld8(vb), ld8(vb + 16));
    }
    const float* pr = S + ln * SROW + j0;
    float4 x0 = *(const float4*)(pr + 8 * h0);
    float4 x1 = *(const float4*)(pr + 8 * h0 + 4);
    float4 x2 = *(const float4*)(pr + 16 + 8 * h0);
    float4 x3 = *(const float4*)(pr + 16 + 8 * h0 + 4);
    v16bf aP;
    aP[0]  = (__bf16)x0.x; aP[1]  = (__bf16)x0.y; aP[2]  = (__bf16)x0.z; aP[3]  = (__bf16)x0.w;
    aP[4]  = (__bf16)x1.x; aP[5]  = (__bf16)x1.y; aP[6]  = (__bf16)x1.z; aP[7]  = (__bf16)x1.w;
    aP[8]  = (__bf16)x2.x; aP[9]  = (__bf16)x2.y; aP[10] = (__bf16)x2.z; aP[11] = (__bf16)x2.w;
    aP[12] = (__bf16)x3.x; aP[13] = (__bf16)x3.y; aP[14] = (__bf16)x3.z; aP[15] = (__bf16)x3.w;
#pragma unroll
    for (int nt = 0; nt < 4; ++nt) acc[nt] = WMMA_BF16(aP, bv[nt], acc[nt]);
  }

  // ---- epilogue: normalize, write in "faithful reshape" order ----
#pragma unroll
  for (int vv = 0; vv < 8; ++vv) {
    const int i    = i0 + vv + 8 * h0;             // sp
    const float nf = 1.0f / rs[vv + 8 * h0];
    const int row  = h * 32 + (i >> 4);            // reshape: no head transpose back
    const int colb = (i & 15) * 64;
#pragma unroll
    for (int nt = 0; nt < 4; ++nt) {
      ao[((size_t)b * 512 + row) * 1024 + colb + nt * 16 + ln] =
          (__bf16)(acc[nt][vv] * nf);
    }
  }
}

// ---------------------------------------------------------------------------
// Kernel 3: output projection [2048,1024] @ [1024,1024] + bias -> f32 out.
// grid = (4 col-groups of 256, 128 row-tiles of 16), block = 128 (4 waves).
// All 4 waves share one 16x1024 bf16 A tile -> stage it once per block in LDS
// via global_load_async_to_lds_b128 (ASYNCcnt), padded rows for conflict-free
// ds_load_b128 fragment reads.
// ---------------------------------------------------------------------------
#define AROW 1032  // bf16 elements per LDS row: 1024 + 8 pad (16B) -> 64-bank spread

__global__ void __launch_bounds__(128)
outproj_kernel(const __bf16* __restrict__ ao, const __bf16* __restrict__ pWo,
               const float* __restrict__ bo, float* __restrict__ out) {
  __shared__ __align__(16) __bf16 Atile[16 * AROW];  // ~32.25 KB
  const int tid  = threadIdx.x;
  const int lane = tid & 31;
  const int w    = tid >> 5;
  const int h0   = lane >> 4;
  const int ln   = lane & 15;
  const int m0      = blockIdx.y * 16;
  const int colbase = blockIdx.x * 256 + w * 64;

  // ---- async DMA: rows m0..m0+15 of ao (contiguous 32 KB) -> LDS ----
  {
    const char* gbase = (const char*)(ao + (size_t)m0 * 1024);
    const unsigned ldsbase =
        (unsigned)(size_t)(__attribute__((address_space(3))) __bf16*)Atile;
#pragma unroll
    for (int row = 0; row < 16; ++row) {
      unsigned    ldsoff = ldsbase + (unsigned)(row * (AROW * 2) + tid * 16);
      const char* g      = gbase + row * 2048 + tid * 16;
      asm volatile("global_load_async_to_lds_b128 %0, %1, off"
                   :: "v"(ldsoff), "v"(g) : "memory");
    }
    asm volatile("s_wait_asynccnt 0x0" ::: "memory");
  }
  __syncthreads();

  v8f acc[4];
#pragma unroll
  for (int i = 0; i < 4; ++i) acc[i] = zero8();

  for (int kt = 0; kt < 32; ++kt) {
    // Preload the four B fragments (global, L2-resident packed weights).
    v16bf bfr[4];
#pragma unroll
    for (int nt4 = 0; nt4 < 4; ++nt4) {
      const int nt = (colbase >> 4) + nt4;
      const __bf16* pf = pWo + (((size_t)(kt * 64 + nt)) * 32 + lane) * 16;
      bfr[nt4] = join8(ld8(pf), ld8(pf + 8));
    }
    // A fragment from LDS (ds_load_b128 x2).
    const __bf16* arow = Atile + ln * AROW + kt * 32;
    v16bf a = join8(ld8(arow + 8 * h0), ld8(arow + 16 + 8 * h0));
#pragma unroll
    for (int nt4 = 0; nt4 < 4; ++nt4) acc[nt4] = WMMA_BF16(a, bfr[nt4], acc[nt4]);
  }

#pragma unroll
  for (int vv = 0; vv < 8; ++vv) {
    const int row = m0 + vv + 8 * h0;
#pragma unroll
    for (int nt4 = 0; nt4 < 4; ++nt4) {
      const int col = colbase + nt4 * 16 + ln;
      out[(size_t)row * 1024 + col] = acc[nt4][vv] + bo[col];
    }
  }
}

// ---------------------------------------------------------------------------
// Host-side launcher (graph-capture safe: stream-ordered kernels only).
// ---------------------------------------------------------------------------
extern "C" void kernel_launch(void* const* d_in, const int* in_sizes, int n_in,
                              void* d_out, int out_size, void* d_ws, size_t ws_size,
                              hipStream_t stream) {
  (void)in_sizes; (void)n_in; (void)out_size; (void)ws_size;

  const float* values = (const float*)d_in[0];
  const float* keys   = (const float*)d_in[1];
  const float* query  = (const float*)d_in[2];
  const int*   mask   = (const int*)d_in[3];
  const float* Wv     = (const float*)d_in[4];
  const float* Wk     = (const float*)d_in[5];
  const float* Wq     = (const float*)d_in[6];
  const float* Wo     = (const float*)d_in[7];
  const float* bo     = (const float*)d_in[8];
  float* out = (float*)d_out;

  // Workspace layout (bf16 elements): qh/kh/vt/ao = 2,097,152 each (4 MB),
  // packed Wq/Wk/Wv = 65,536 each, packed Wo = 1,048,576.  Total ~18.4 MB.
  __bf16* qh  = (__bf16*)d_ws;
  __bf16* kh  = qh  + (size_t)2097152;
  __bf16* vt  = kh  + (size_t)2097152;
  __bf16* ao  = vt  + (size_t)2097152;
  __bf16* pWq = ao  + (size_t)2097152;
  __bf16* pWk = pWq + (size_t)65536;
  __bf16* pWv = pWk + (size_t)65536;
  __bf16* pWo = pWv + (size_t)65536;

  pack_w_kernel<<<64,  256, 0, stream>>>(Wq, pWq, 1024, 64);
  pack_w_kernel<<<64,  256, 0, stream>>>(Wk, pWk, 1024, 64);
  pack_w_kernel<<<64,  256, 0, stream>>>(Wv, pWv, 1024, 64);
  pack_w_kernel<<<512, 256, 0, stream>>>(Wo, pWo, 1024, 1024);

  proj_kernel<<<512, 128, 0, stream>>>(query,  pWq, qh, 0);
  proj_kernel<<<512, 128, 0, stream>>>(keys,   pWk, kh, 0);
  proj_kernel<<<512, 128, 0, stream>>>(values, pWv, vt, 1);

  const size_t smem = (size_t)(16 * SROW + 16) * sizeof(float);  // 33 KB
  attn_kernel<<<dim3(32, 64), 32, smem, stream>>>(qh, kh, vt, mask, ao);

  outproj_kernel<<<dim3(4, 128), 128, 0, stream>>>(ao, pWo, bo, out);
}
